// DeepMultiBasisBilinearNet_26225070309871
// MI455X (gfx1250) — matrix operand order repaired and checked by reading, not compiled, not run on hardware
//
#include <hip/hip_runtime.h>

// ---------------------------------------------------------------------------
// DeepMultiBasisBilinearNet on MI455X (gfx1250, wave32).
//
// Compute-bound fp32 net (~825 GFLOP vs ~1 GB HBM -> memory is ~free).
// GEMMs use split-bf16 ("3xBF16") on v_wmma_f32_16x16x32_bf16:
//   x = xh + xl;  A*B ~= Ah*Bh + Ah*Bl + Al*Bh   (fp32 accumulate, ~1e-5 rel)
// Tiles are double-buffered in LDS and streamed with CDNA5 async-to-LDS
// copies (GLOBAL_LOAD_ASYNC_TO_LDS_*, ASYNCcnt) so HBM/L2 latency overlaps
// the WMMA stream (bf16 WMMA co-executes with VALU as a TRANS op).
//
// Pipeline:
//   k_split     : weights -> persistent bf16 hi/lo planes (once per call)
//   k_bilinear  : interH/L = split_bf16((X@Wr^T+br)*(X@Wl^T+bl))   [B,HR]
//   k_expand_ln : H = LN(X + inter@We^T + be)*g + b                [B,D]
//   k_head      : out = H@Wf^T + bf (trivial VALU)                 [B,10]
// ---------------------------------------------------------------------------

#define B_ROWS 16384
#define D_IN   1024
#define HR     4096
#define N_OUT  10

typedef int    v4i   __attribute__((ext_vector_type(4)));
typedef int    v8i   __attribute__((ext_vector_type(8)));
typedef float  v8f   __attribute__((ext_vector_type(8)));
typedef __bf16 v16bf __attribute__((ext_vector_type(16)));

// gcc-style vector types matching the async-to-LDS builtin signatures
typedef int gv4i __attribute__((vector_size(16)));
typedef int gv2i __attribute__((vector_size(8)));

static __device__ inline void split1(float x, unsigned short& h, unsigned short& l) {
  __bf16 hb = (__bf16)x;
  __bf16 lb = (__bf16)(x - (float)hb);
  h = __builtin_bit_cast(unsigned short, hb);
  l = __builtin_bit_cast(unsigned short, lb);
}
static __device__ inline unsigned int pk(unsigned short a, unsigned short b) {
  return (unsigned int)a | ((unsigned int)b << 16);
}
static __device__ inline v4i ld4s(const unsigned int* p) { return *(const v4i*)p; }
static __device__ inline v16bf mkfrag(v4i a, v4i b) {
  v8i w = __builtin_shufflevector(a, b, 0, 1, 2, 3, 4, 5, 6, 7);
  return __builtin_bit_cast(v16bf, w);
}
#define WMMA_BF16(A, Bm, C) \
  __builtin_amdgcn_wmma_f32_16x16x32_bf16(false, (A), false, (Bm), (short)0, (C), false, false)

// --- CDNA5 async global->LDS copies (ASYNCcnt) -----------------------------
static __device__ inline void async_b128(const unsigned int* g, unsigned int* l) {
  __builtin_amdgcn_global_load_async_to_lds_b128(
      (__attribute__((address_space(1))) gv4i*)g,
      (__attribute__((address_space(3))) gv4i*)l, 0, 0);
}
static __device__ inline void async_b64(const unsigned int* g, unsigned int* l) {
  __builtin_amdgcn_global_load_async_to_lds_b64(
      (__attribute__((address_space(1))) gv2i*)g,
      (__attribute__((address_space(3))) gv2i*)l, 0, 0);
}
static __device__ inline void wait_async0() {
#if __has_builtin(__builtin_amdgcn_s_wait_asynccnt)
  __builtin_amdgcn_s_wait_asynccnt(0);
#else
  asm volatile("s_wait_asynccnt 0x0" ::: "memory");
#endif
}

// ---------------------------------------------------------------------------
// Weight splitter: fp32 [N*K] -> bf16 hi plane + bf16 lo plane (pair-dwords,
// even-K element in [15:0] to match 16-bit WMMA VGPR packing).
// ---------------------------------------------------------------------------
__global__ __launch_bounds__(256) void k_split(
    const float* __restrict__ W,
    unsigned int* __restrict__ hi, unsigned int* __restrict__ lo, int nquads)
{
  const int q = blockIdx.x * 256 + threadIdx.x;
  if (q >= nquads) return;
  const float4 v = ((const float4*)W)[q];
  unsigned short h0, l0, h1, l1, h2, l2, h3, l3;
  split1(v.x, h0, l0); split1(v.y, h1, l1);
  split1(v.z, h2, l2); split1(v.w, h3, l3);
  ((uint2*)hi)[q] = make_uint2(pk(h0, h1), pk(h2, h3));
  ((uint2*)lo)[q] = make_uint2(pk(l0, l1), pk(l2, l3));
}

// ---------------------------------------------------------------------------
// Kernel 1: bilinear interaction pair, split-bf16, async double-buffered.
// Grid (16, 1024), block 128 (4 waves). WG: 16(M) x 256(N) of both R and L.
// LDS (dynamic, dwords): A[2 buf][2 planes][320] | B[2 buf][4 planes][4096]
// ---------------------------------------------------------------------------
constexpr int KT = 32;

__global__ __launch_bounds__(128) void k_bilinear(
    const float* __restrict__ X,
    const unsigned int* __restrict__ WrH, const unsigned int* __restrict__ WrL,
    const unsigned int* __restrict__ WlH, const unsigned int* __restrict__ WlL,
    const float* __restrict__ br, const float* __restrict__ bl,
    unsigned short* __restrict__ interH, unsigned short* __restrict__ interL)
{
  extern __shared__ unsigned int smem_u[];
  unsigned int* sA = smem_u;             // 2*640 dwords
  unsigned int* sB = smem_u + 1280;      // 2*16384 dwords

  const int tid  = threadIdx.x;
  const int lane = tid & 31;
  const int wave = tid >> 5;             // 0..3
  const int hl   = lane >> 4;
  const int lm   = lane & 15;
  const int m0   = blockIdx.y * 16;
  const int n0wg = blockIdx.x * 256;

  const unsigned int* gB[4] = {WrH, WrL, WlH, WlL};

  v8f accR[4] = {};
  v8f accL[4] = {};

  auto stageB = [&](int k0, int buf) {
    #pragma unroll
    for (int p = 0; p < 4; ++p) {
      const unsigned int* gp = gB[p];
      unsigned int* sp = sB + buf * 16384 + p * 4096;
      #pragma unroll
      for (int i = 0; i < 8; ++i) {
        const int idx = tid + (i << 7);      // 0..1023
        const int n = idx >> 2, c = idx & 3;
        async_b128(&gp[(size_t)(n0wg + n) * (D_IN / 2) + (k0 >> 1) + (c << 2)],
                   &sp[n * 16 + ((c ^ ((n >> 2) & 3)) << 2)]);
      }
    }
  };
  auto loadA = [&](int k0) -> float4 {
    const int m = tid >> 3, seg = tid & 7;
    return *(const float4*)&X[(size_t)(m0 + m) * D_IN + k0 + seg * 4];
  };
  auto storeA = [&](float4 xv, int buf) {
    const int m = tid >> 3, seg = tid & 7;
    unsigned int* aH = sA + buf * 640;
    unsigned int* aL = aH + 320;
    unsigned short h0, l0, h1, l1, h2, l2, h3, l3;
    split1(xv.x, h0, l0); split1(xv.y, h1, l1);
    split1(xv.z, h2, l2); split1(xv.w, h3, l3);
    aH[m * 20 + seg * 2]     = pk(h0, h1);
    aH[m * 20 + seg * 2 + 1] = pk(h2, h3);
    aL[m * 20 + seg * 2]     = pk(l0, l1);
    aL[m * 20 + seg * 2 + 1] = pk(l2, l3);
  };

  // prologue: fill buffer 0
  {
    float4 xv = loadA(0);
    stageB(0, 0);
    storeA(xv, 0);
  }
  wait_async0();
  __syncthreads();

  for (int k0 = 0; k0 < D_IN; k0 += KT) {
    const int  buf  = (k0 >> 5) & 1;
    const bool more = (k0 + KT) < D_IN;
    float4 xv = make_float4(0.f, 0.f, 0.f, 0.f);
    if (more) {                       // issue next chunk while computing
      xv = loadA(k0 + KT);
      stageB(k0 + KT, buf ^ 1);
    }

    const unsigned int* aH = sA + buf * 640;
    const unsigned int* aL = aH + 320;
    const unsigned int* bb = sB + buf * 16384;
    const v16bf Ah = mkfrag(ld4s(&aH[lm * 20 + 4 * hl]),
                            ld4s(&aH[lm * 20 + 8 + 4 * hl]));
    const v16bf Al = mkfrag(ld4s(&aL[lm * 20 + 4 * hl]),
                            ld4s(&aL[lm * 20 + 8 + 4 * hl]));
    #pragma unroll
    for (int t = 0; t < 4; ++t) {
      const int n  = wave * 64 + t * 16 + lm;
      const int sw = (n >> 2) & 3;
      const int o0 = n * 16 + (((hl)     ^ sw) << 2);
      const int o1 = n * 16 + (((hl + 2) ^ sw) << 2);
      const v16bf BrH = mkfrag(ld4s(&bb[0 * 4096 + o0]), ld4s(&bb[0 * 4096 + o1]));
      const v16bf BrL = mkfrag(ld4s(&bb[1 * 4096 + o0]), ld4s(&bb[1 * 4096 + o1]));
      const v16bf BlH = mkfrag(ld4s(&bb[2 * 4096 + o0]), ld4s(&bb[2 * 4096 + o1]));
      const v16bf BlL = mkfrag(ld4s(&bb[3 * 4096 + o0]), ld4s(&bb[3 * 4096 + o1]));
      accR[t] = WMMA_BF16(Ah, BrH, accR[t]);
      accR[t] = WMMA_BF16(Ah, BrL, accR[t]);
      accR[t] = WMMA_BF16(Al, BrH, accR[t]);
      accL[t] = WMMA_BF16(Ah, BlH, accL[t]);
      accL[t] = WMMA_BF16(Ah, BlL, accL[t]);
      accL[t] = WMMA_BF16(Al, BlH, accL[t]);
    }

    if (more) storeA(xv, buf ^ 1);    // convert after compute (hide latency)
    wait_async0();
    __syncthreads();
  }

  // epilogue: inter = (R+br)*(L+bl), stored as bf16 hi/lo planes
  #pragma unroll
  for (int t = 0; t < 4; ++t) {
    const int n = n0wg + wave * 64 + t * 16 + lm;
    const float brv = br[n], blv = bl[n];
    #pragma unroll
    for (int r = 0; r < 8; ++r) {
      const int m = m0 + r + 8 * hl;
      const float v = (accR[t][r] + brv) * (accL[t][r] + blv);
      unsigned short h, l;
      split1(v, h, l);
      interH[(size_t)m * HR + n] = h;
      interL[(size_t)m * HR + n] = l;
    }
  }
}

// ---------------------------------------------------------------------------
// Kernel 2: expand GEMM + residual + LayerNorm, split-bf16, async dbl-buffer.
// Grid 1024, block 256 (8 waves). WG owns all N=1024 for 16 rows -> LN in-WG.
// LDS (dynamic, dwords): A[2 buf][2 planes][320] | B[2 buf][2 planes][16384]
// ---------------------------------------------------------------------------
__global__ __launch_bounds__(256) void k_expand_ln(
    const unsigned int* __restrict__ iH, const unsigned int* __restrict__ iL,
    const float* __restrict__ Xres,
    const unsigned int* __restrict__ WeH, const unsigned int* __restrict__ WeL,
    const float* __restrict__ be, const float* __restrict__ gam,
    const float* __restrict__ bet, float* __restrict__ Hout)
{
  extern __shared__ unsigned int smem_u[];
  unsigned int* sA = smem_u;             // 2*640 dwords
  unsigned int* sB = smem_u + 1280;      // 2*32768 dwords
  __shared__ float rsum[16];
  __shared__ float rsumsq[16];

  const int tid  = threadIdx.x;
  const int lane = tid & 31;
  const int wave = tid >> 5;             // 0..7
  const int hl   = lane >> 4;
  const int lm   = lane & 15;
  const int m0   = blockIdx.x * 16;

  v8f acc[8] = {};

  auto stageA = [&](int k0, int buf) {
    const int p  = tid >> 7;             // plane
    const int r  = (tid >> 3) & 15;
    const int pq = (tid & 7) * 2;
    const unsigned int* ap = p ? iL : iH;
    unsigned int* sp = sA + buf * 640 + p * 320;
    async_b64(&ap[(size_t)(m0 + r) * (HR / 2) + (k0 >> 1) + pq], &sp[r * 20 + pq]);
  };
  auto stageB = [&](int k0, int buf) {
    #pragma unroll
    for (int p = 0; p < 2; ++p) {
      const unsigned int* gp = p ? WeL : WeH;
      unsigned int* sp = sB + buf * 32768 + p * 16384;
      #pragma unroll
      for (int i = 0; i < 16; ++i) {
        const int idx = tid + (i << 8);  // 0..4095
        const int n = idx >> 2, c = idx & 3;
        async_b128(&gp[(size_t)n * (HR / 2) + (k0 >> 1) + (c << 2)],
                   &sp[n * 16 + ((c ^ ((n >> 2) & 3)) << 2)]);
      }
    }
  };

  stageA(0, 0);
  stageB(0, 0);
  wait_async0();
  __syncthreads();

  for (int k0 = 0; k0 < HR; k0 += KT) {
    const int  buf  = (k0 >> 5) & 1;
    const bool more = (k0 + KT) < HR;
    if (more) {
      stageA(k0 + KT, buf ^ 1);
      stageB(k0 + KT, buf ^ 1);
    }

    const unsigned int* aH = sA + buf * 640;
    const unsigned int* aL = aH + 320;
    const unsigned int* bb = sB + buf * 32768;
    const v16bf Ah = mkfrag(ld4s(&aH[lm * 20 + 4 * hl]),
                            ld4s(&aH[lm * 20 + 8 + 4 * hl]));
    const v16bf Al = mkfrag(ld4s(&aL[lm * 20 + 4 * hl]),
                            ld4s(&aL[lm * 20 + 8 + 4 * hl]));
    #pragma unroll
    for (int t = 0; t < 8; ++t) {
      const int n  = wave * 128 + t * 16 + lm;
      const int sw = (n >> 2) & 3;
      const int o0 = n * 16 + (((hl)     ^ sw) << 2);
      const int o1 = n * 16 + (((hl + 2) ^ sw) << 2);
      const v16bf Bh = mkfrag(ld4s(&bb[o0]),         ld4s(&bb[o1]));
      const v16bf Bl = mkfrag(ld4s(&bb[16384 + o0]), ld4s(&bb[16384 + o1]));
      acc[t] = WMMA_BF16(Ah, Bh, acc[t]);
      acc[t] = WMMA_BF16(Ah, Bl, acc[t]);
      acc[t] = WMMA_BF16(Al, Bh, acc[t]);
    }

    wait_async0();
    __syncthreads();
  }

  // epilogue: y = acc + be + residual, then LayerNorm over n=0..1023
  if (tid < 16) { rsum[tid] = 0.0f; rsumsq[tid] = 0.0f; }
  __syncthreads();

  float psum[8], psumsq[8];
  #pragma unroll
  for (int r = 0; r < 8; ++r) { psum[r] = 0.0f; psumsq[r] = 0.0f; }

  #pragma unroll
  for (int t = 0; t < 8; ++t) {
    const int n = wave * 128 + t * 16 + lm;
    const float bev = be[n];
    #pragma unroll
    for (int r = 0; r < 8; ++r) {
      const int m = m0 + r + 8 * hl;
      const float y = acc[t][r] + bev + Xres[(size_t)m * D_IN + n];
      acc[t][r] = y;
      psum[r]   += y;
      psumsq[r] += y * y;
    }
  }
  #pragma unroll
  for (int r = 0; r < 8; ++r) {
    #pragma unroll
    for (int off = 8; off >= 1; off >>= 1) {
      psum[r]   += __shfl_xor(psum[r],   off, 32);
      psumsq[r] += __shfl_xor(psumsq[r], off, 32);
    }
  }
  if (lm == 0) {
    #pragma unroll
    for (int r = 0; r < 8; ++r) {
      atomicAdd(&rsum  [r + 8 * hl], psum[r]);
      atomicAdd(&rsumsq[r + 8 * hl], psumsq[r]);
    }
  }
  __syncthreads();

  #pragma unroll
  for (int t = 0; t < 8; ++t) {
    const int n = wave * 128 + t * 16 + lm;
    const float gv = gam[n], bv = bet[n];
    #pragma unroll
    for (int r = 0; r < 8; ++r) {
      const int m = r + 8 * hl;
      const float mu  = rsum[m]   * (1.0f / (float)D_IN);
      const float var = rsumsq[m] * (1.0f / (float)D_IN) - mu * mu;
      const float rs  = __frsqrt_rn(var + 1e-5f);
      Hout[(size_t)(m0 + m) * D_IN + n] = (acc[t][r] - mu) * rs * gv + bv;
    }
  }
}

// ---------------------------------------------------------------------------
// Kernel 3: final linear head (N=10), plain VALU (0.04% of FLOPs).
// ---------------------------------------------------------------------------
__global__ __launch_bounds__(256) void k_head(
    const float* __restrict__ H, const float* __restrict__ Wf,
    const float* __restrict__ bf, float* __restrict__ out)
{
  __shared__ float wfs[N_OUT][D_IN];
  const int tid = threadIdx.x;
  for (int i = tid; i < N_OUT * D_IN / 4; i += 256)
    ((float4*)&wfs[0][0])[i] = ((const float4*)Wf)[i];
  __syncthreads();

  const int lane = tid & 31;
  const int wave = tid >> 5;
  const int b    = blockIdx.x * 8 + wave;
  const float* hrow = H + (size_t)b * D_IN;

  float acc[N_OUT];
  #pragma unroll
  for (int o = 0; o < N_OUT; ++o) acc[o] = 0.0f;

  #pragma unroll
  for (int j = 0; j < 8; ++j) {
    const int c = (lane + j * 32) * 4;
    const float4 hv = *(const float4*)&hrow[c];
    #pragma unroll
    for (int o = 0; o < N_OUT; ++o)
      acc[o] += hv.x * wfs[o][c]     + hv.y * wfs[o][c + 1]
              + hv.z * wfs[o][c + 2] + hv.w * wfs[o][c + 3];
  }
  #pragma unroll
  for (int o = 0; o < N_OUT; ++o) {
    #pragma unroll
    for (int off = 16; off >= 1; off >>= 1)
      acc[o] += __shfl_xor(acc[o], off, 32);
  }
  if (lane == 0) {
    #pragma unroll
    for (int o = 0; o < N_OUT; ++o)
      out[(size_t)b * N_OUT + o] = acc[o] + bf[o];
  }
}

// ---------------------------------------------------------------------------
extern "C" void kernel_launch(void* const* d_in, const int* in_sizes, int n_in,
                              void* d_out, int out_size, void* d_ws, size_t ws_size,
                              hipStream_t stream) {
  const float* x   = (const float*)d_in[0];
  const float* wr1 = (const float*)d_in[1];
  const float* br1 = (const float*)d_in[2];
  const float* wl1 = (const float*)d_in[3];
  const float* bl1 = (const float*)d_in[4];
  const float* we1 = (const float*)d_in[5];
  const float* be1 = (const float*)d_in[6];
  const float* g1  = (const float*)d_in[7];
  const float* b1  = (const float*)d_in[8];
  const float* wr2 = (const float*)d_in[9];
  const float* br2 = (const float*)d_in[10];
  const float* wl2 = (const float*)d_in[11];
  const float* bl2 = (const float*)d_in[12];
  const float* we2 = (const float*)d_in[13];
  const float* be2 = (const float*)d_in[14];
  const float* g2  = (const float*)d_in[15];
  const float* b2  = (const float*)d_in[16];
  const float* wf  = (const float*)d_in[17];
  const float* bf  = (const float*)d_in[18];

  size_t off = 0;
  auto ws_alloc = [&](size_t bytes) -> void* {
    void* p = (void*)((char*)d_ws + off);
    off += (bytes + 255) & ~(size_t)255;
    return p;
  };

  unsigned short* interH = (unsigned short*)ws_alloc((size_t)B_ROWS * HR * 2);
  unsigned short* interL = (unsigned short*)ws_alloc((size_t)B_ROWS * HR * 2);
  float* h1 = (float*)ws_alloc((size_t)B_ROWS * D_IN * 4);
  float* h2 = (float*)ws_alloc((size_t)B_ROWS * D_IN * 4);

  const float* wsrc[6] = {wr1, wl1, we1, wr2, wl2, we2};
  unsigned int* WH[6];
  unsigned int* WL[6];
  const size_t welems = (size_t)HR * D_IN;          // 4M elements each
  for (int i = 0; i < 6; ++i) {
    WH[i] = (unsigned int*)ws_alloc(welems * 2);
    WL[i] = (unsigned int*)ws_alloc(welems * 2);
  }
  const int nquads = (int)(welems / 4);
  for (int i = 0; i < 6; ++i)
    k_split<<<nquads / 256, 256, 0, stream>>>(wsrc[i], WH[i], WL[i], nquads);

  const dim3 grid1(HR / 256, B_ROWS / 16);          // (16, 1024)
  const size_t smem1 = (size_t)(1280 + 2 * 4 * 4096) * 4;   // 136,192 B
  const size_t smem2 = (size_t)(1280 + 2 * 2 * 16384) * 4;  // 267,264 B

  k_bilinear<<<grid1, 128, smem1, stream>>>(x, WH[0], WL[0], WH[1], WL[1],
                                            br1, bl1, interH, interL);
  k_expand_ln<<<B_ROWS / 16, 256, smem2, stream>>>(
      (const unsigned int*)interH, (const unsigned int*)interL, x,
      WH[2], WL[2], be1, g1, b1, h1);
  k_bilinear<<<grid1, 128, smem1, stream>>>(h1, WH[3], WL[3], WH[4], WL[4],
                                            br2, bl2, interH, interL);
  k_expand_ln<<<B_ROWS / 16, 256, smem2, stream>>>(
      (const unsigned int*)interH, (const unsigned int*)interL, h1,
      WH[5], WL[5], be2, g2, b2, h2);
  k_head<<<B_ROWS / 8, 256, 0, stream>>>(h2, wf, bf, (float*)d_out);
}
